// LlamaAttention_3702261809776
// MI455X (gfx1250) — matrix operand order repaired and checked by reading, not compiled, not run on hardware
//
#include <hip/hip_runtime.h>
#include <hip/hip_bf16.h>

typedef __attribute__((ext_vector_type(16))) __bf16 v16bf;
typedef __attribute__((ext_vector_type(8)))  __bf16 v8bf;
typedef __attribute__((ext_vector_type(4)))  __bf16 v4bf;
typedef __attribute__((ext_vector_type(8)))  float  v8f;

#define HIDDEN   4096
#define NHEADS   32
#define HDIM     128
#define SEQ      2048
#define BATCH    2

__device__ __forceinline__ v8f wmma_bf16(v16bf a, v16bf b, v8f c) {
    // D = A(16x32) * B(32x16) + C, f32 accumulate
    return __builtin_amdgcn_wmma_f32_16x16x32_bf16(
        /*neg_a=*/false, a, /*neg_b=*/false, b,
        /*c_mod=*/(short)0, c, /*reuse_a=*/false, /*reuse_b=*/false);
}

// Async global->LDS copy of 16 bytes (ASYNCcnt tracked, bypasses VGPRs).
// LDS offset = low 32 bits of the flat address.
__device__ __forceinline__ unsigned lds_off(const void* p) {
    return (unsigned)(unsigned long long)p;
}
__device__ __forceinline__ void async_copy16(const void* gsrc, unsigned ldst) {
    asm volatile("global_load_async_to_lds_b128 %0, %1, off"
                 :: "v"(ldst), "v"(gsrc) : "memory");
}
__device__ __forceinline__ void wait_async0() {
    asm volatile("s_wait_asynccnt 0" ::: "memory");
}

// ---------------------------------------------------------------- convert
__global__ __launch_bounds__(256)
void cvt_f32_bf16(const float4* __restrict__ in, __bf16* __restrict__ out, int n4) {
    int i = blockIdx.x * blockDim.x + threadIdx.x;
    if (i < n4) {
        float4 v = in[i];
        v4bf o;
        o[0] = (__bf16)v.x; o[1] = (__bf16)v.y;
        o[2] = (__bf16)v.z; o[3] = (__bf16)v.w;
        *(v4bf*)(out + (size_t)i * 4) = o;
    }
}

// ---------------------------------------------------------------- GEMM
// C[M,N] = A[M,K] * B[K,N], bf16 in, f32 accumulate, bf16 or f32 out.
// 256 thr (8 waves), tile 128x128, K-step 32, double-buffered LDS:
//   A tile: async global->LDS (ASYNCcnt), B tile: pipelined VGPR transpose.
template<bool OUT_F32>
__global__ __launch_bounds__(256)
void gemm_bf16_kernel(const __bf16* __restrict__ A, const __bf16* __restrict__ B,
                      void* __restrict__ Cout, int M, int N, int K)
{
    __shared__ __attribute__((aligned(32))) __bf16 lA[2][128 * 32];   // [row][k]
    __shared__ __attribute__((aligned(32))) __bf16 lBt[2][128 * 32];  // [n][k]

    const int t    = threadIdx.x;
    const int lane = t & 31;
    const int wave = t >> 5;
    const int lo   = lane & 15;
    const int hi   = lane >> 4;
    const int rowBase = blockIdx.y * 128;
    const int colBase = blockIdx.x * 128;

    const int rA = t >> 1, khA = t & 1;        // A staging: 32B per thread
    const int kB = t >> 3, ncB = t & 7;        // B staging: 32B per thread

    auto stageA = [&](int kb, int b) {
        const __bf16* src = A + (size_t)(rowBase + rA) * K + kb + khA * 16;
        unsigned dst = lds_off(&lA[b][rA * 32 + khA * 16]);
        async_copy16(src, dst);
        async_copy16(src + 8, dst + 16);
    };
    auto loadB = [&](int kb) -> v16bf {
        return *(const v16bf*)(B + (size_t)(kb + kB) * N + colBase + ncB * 16);
    };
    auto scatterB = [&](v16bf v, int b) {
        #pragma unroll
        for (int j = 0; j < 16; j++) lBt[b][(ncB * 16 + j) * 32 + kB] = v[j];
    };

    v8f acc[8] = {};
    const int steps = K >> 5;

    // prologue: stage tile 0
    stageA(0, 0);
    { v16bf b0 = loadB(0); scatterB(b0, 0); }
    wait_async0();
    __syncthreads();

    int buf = 0;
    for (int s = 0; s < steps; s++) {
        const int kb = s * 32;
        const bool hasNext = (s + 1 < steps);
        v16bf bv;
        if (hasNext) {
            stageA(kb + 32, buf ^ 1);          // async fill of next buffer
            bv = loadB(kb + 32);               // B regs for next buffer
        }
        if (s + 2 < steps)                     // L2 warm for step s+2
            __builtin_prefetch(B + (size_t)(kb + 64 + kB) * N + colBase + ncB * 16, 0, 0);

        // A fragment: lane = row (wave*16+lo); contiguous K split, k = 16*hi + s
        v16bf afrag;
        {
            const __bf16* ap = &lA[buf][(wave * 16 + lo) * 32 + (hi ? 16 : 0)];
            v8bf a0 = *(const v8bf*)ap;
            v8bf a1 = *(const v8bf*)(ap + 8);
            #pragma unroll
            for (int j = 0; j < 8; j++) { afrag[j] = a0[j]; afrag[8 + j] = a1[j]; }
        }
        #pragma unroll
        for (int c = 0; c < 8; c++) {
            v16bf bfrag = *(const v16bf*)&lBt[buf][(c * 16 + lo) * 32 + (hi ? 16 : 0)];
            acc[c] = wmma_bf16(afrag, bfrag, acc[c]);
        }

        if (hasNext) scatterB(bv, buf ^ 1);    // fill next B buffer behind WMMAs
        wait_async0();
        __syncthreads();
        buf ^= 1;
    }

    // C/D layout: VGPR j <-> row j + 8*hi, column = lane&15
    #pragma unroll
    for (int c = 0; c < 8; c++) {
        int col = colBase + c * 16 + lo;
        #pragma unroll
        for (int j = 0; j < 8; j++) {
            int row = rowBase + wave * 16 + j + hi * 8;
            if (OUT_F32) ((float*)Cout)[(size_t)row * N + col] = acc[c][j];
            else ((__bf16*)Cout)[(size_t)row * N + col] = (__bf16)acc[c][j];
        }
    }
}

// ---------------------------------------------------------------- RoPE + relayout
// In : Qr/Kr/Vr bf16 [B,S,H*D]; Out: Qa/Ka bf16 [B,H,S,D] roped, Vt bf16 [B,H,D,S]
__global__ __launch_bounds__(256)
void rope_kernel(const int* __restrict__ pos,
                 const __bf16* __restrict__ Qr, const __bf16* __restrict__ Kr,
                 const __bf16* __restrict__ Vr,
                 __bf16* __restrict__ Qa, __bf16* __restrict__ Ka,
                 __bf16* __restrict__ Vt)
{
    int tid = blockIdx.x * blockDim.x + threadIdx.x;   // 2^23 threads
    int dp = tid & 63;
    int h  = (tid >> 6) & 31;
    int s  = (tid >> 11) & 2047;
    int b  = tid >> 22;

    float p   = (float)pos[s];
    float inv = __powf(10000.0f, -((float)(2 * dp)) * (1.0f / 128.0f));
    float c, sn;
    __sincosf(p * inv, &sn, &c);

    size_t src = ((size_t)(b * SEQ + s)) * HIDDEN + (size_t)h * HDIM;
    float q0 = (float)Qr[src + dp], q1 = (float)Qr[src + 64 + dp];
    float k0 = (float)Kr[src + dp], k1 = (float)Kr[src + 64 + dp];
    float v0 = (float)Vr[src + dp], v1 = (float)Vr[src + 64 + dp];

    size_t dst = ((size_t)(b * NHEADS + h) * SEQ + s) * HDIM;
    Qa[dst + dp]      = (__bf16)(q0 * c - q1 * sn);
    Qa[dst + 64 + dp] = (__bf16)(q1 * c + q0 * sn);
    Ka[dst + dp]      = (__bf16)(k0 * c - k1 * sn);
    Ka[dst + 64 + dp] = (__bf16)(k1 * c + k0 * sn);

    size_t vb = (size_t)(b * NHEADS + h) * HDIM;
    Vt[(vb + dp) * SEQ + s]      = (__bf16)v0;
    Vt[(vb + 64 + dp) * SEQ + s] = (__bf16)v1;
}

// ---------------------------------------------------------------- Flash attention
// Q,K: [B,H,S,D] bf16 (roped); Vt: [B,H,D,S] bf16; out: [B,S,H*D] bf16.
// 1024 blocks x 256 thr. All 8 waves of a block share (b,h): K/V tiles are
// staged once per block into LDS via async-to-LDS copies.
__global__ __launch_bounds__(256)
void attn_kernel(const __bf16* __restrict__ Q, const __bf16* __restrict__ Kk,
                 const __bf16* __restrict__ Vt, __bf16* __restrict__ Oout)
{
    __shared__ __attribute__((aligned(32))) __bf16 lK[32 * 128];   // [key][d]
    __shared__ __attribute__((aligned(32))) __bf16 lV[128 * 32];   // [d][k]
    __shared__ __attribute__((aligned(32))) __bf16 lP[8][16 * 32]; // per-wave P

    const int t    = threadIdx.x;
    const int lane = t & 31;
    const int wave = t >> 5;
    const int lo   = lane & 15;
    const int hi   = lane >> 4;

    int bid = blockIdx.x;
    int qc  = bid & 15;
    int h   = (bid >> 4) & 31;
    int b   = bid >> 9;
    int qbase = qc * 128 + wave * 16;
    const float scale = 0.08838834764831845f;   // 1/sqrt(128)

    size_t bh = (size_t)(b * NHEADS + h);
    const __bf16* Qp = Q  + (bh * SEQ + qbase) * HDIM;
    const __bf16* Kp = Kk +  bh * SEQ * HDIM;
    const __bf16* Vp = Vt +  bh * HDIM * SEQ;

    // Q as 4 A-fragments (d = kd*32 .. +31)
    v16bf qf[4];
    #pragma unroll
    for (int kd = 0; kd < 4; kd++) {
        const __bf16* p = Qp + (size_t)lo * HDIM + kd * 32 + (hi ? 16 : 0);
        v8bf a0 = *(const v8bf*)p;
        v8bf a1 = *(const v8bf*)(p + 8);
        #pragma unroll
        for (int j = 0; j < 8; j++) { qf[kd][j] = a0[j]; qf[kd][8 + j] = a1[j]; }
    }

    v8f O[8] = {};
    float mrow[8], lrow[8];
    #pragma unroll
    for (int j = 0; j < 8; j++) { mrow[j] = -3.0e38f; lrow[j] = 0.0f; }

    const int myNkt  = (qbase + 16 + 31) >> 5;        // causal extent of this wave
    const int blkNkt = (qc * 128 + 128 + 31) >> 5;    // max extent in block

    const int kKey = t >> 3, kSeg = t & 7;            // K staging map
    const int vD   = t >> 1, vHalf = t & 1;           // V staging map

    for (int kt = 0; kt < blkNkt; kt++) {
        const int kbase = kt * 32;

        // cooperative async staging of K (32x128) and V^T (128x32) tiles
        {
            const __bf16* ks = Kp + (size_t)(kbase + kKey) * HDIM + kSeg * 16;
            unsigned kd = lds_off(&lK[kKey * 128 + kSeg * 16]);
            async_copy16(ks, kd);
            async_copy16(ks + 8, kd + 16);
            const __bf16* vs = Vp + (size_t)vD * SEQ + kbase + vHalf * 16;
            unsigned vd = lds_off(&lV[vD * 32 + vHalf * 16]);
            async_copy16(vs, vd);
            async_copy16(vs + 8, vd + 16);
        }
        wait_async0();
        __syncthreads();

        if (kt < myNkt) {
            v8f S0 = {}, S1 = {};
            #pragma unroll
            for (int kd = 0; kd < 4; kd++) {
                v16bf kf0 = *(const v16bf*)&lK[(lo)      * 128 + kd * 32 + (hi ? 16 : 0)];
                v16bf kf1 = *(const v16bf*)&lK[(16 + lo) * 128 + kd * 32 + (hi ? 16 : 0)];
                S0 = wmma_bf16(qf[kd], kf0, S0);
                S1 = wmma_bf16(qf[kd], kf1, S1);
            }

            __bf16* pw = lP[wave];
            #pragma unroll
            for (int j = 0; j < 8; j++) {
                int M = j + hi * 8;
                int qrow = qbase + M;
                float s0 = S0[j] * scale;
                float s1 = S1[j] * scale;
                if (kbase + lo > qrow)      s0 = -3.0e38f;
                if (kbase + 16 + lo > qrow) s1 = -3.0e38f;
                float mx = fmaxf(s0, s1);
                #pragma unroll
                for (int off = 1; off < 16; off <<= 1)
                    mx = fmaxf(mx, __shfl_xor(mx, off, 32));
                float mnew  = fmaxf(mrow[j], mx);
                float alpha = __expf(mrow[j] - mnew);
                float p0 = __expf(s0 - mnew);
                float p1 = __expf(s1 - mnew);
                float rs = p0 + p1;
                #pragma unroll
                for (int off = 1; off < 16; off <<= 1)
                    rs += __shfl_xor(rs, off, 32);
                lrow[j] = lrow[j] * alpha + rs;
                mrow[j] = mnew;
                #pragma unroll
                for (int c = 0; c < 8; c++) O[c][j] *= alpha;
                pw[M * 32 + lo]      = (__bf16)p0;
                pw[M * 32 + 16 + lo] = (__bf16)p1;
            }
            // per-wave LDS RAW: ds stores must land before fragment reload
            asm volatile("s_wait_dscnt 0" ::: "memory");

            v16bf pf;
            {
                const __bf16* pp = pw + lo * 32 + (hi ? 16 : 0);
                v8bf a0 = *(const v8bf*)pp;
                v8bf a1 = *(const v8bf*)(pp + 8);
                #pragma unroll
                for (int j = 0; j < 8; j++) { pf[j] = a0[j]; pf[8 + j] = a1[j]; }
            }
            #pragma unroll
            for (int dt = 0; dt < 8; dt++) {
                v16bf vf = *(const v16bf*)&lV[(dt * 16 + lo) * 32 + (hi ? 16 : 0)];
                O[dt] = wmma_bf16(pf, vf, O[dt]);
            }
        }
        __syncthreads();   // tiles consumed before next async fill
    }

    // epilogue: normalize + scatter to [B,S,H*D]
    #pragma unroll
    for (int j = 0; j < 8; j++) {
        int M = j + hi * 8;
        int qrow = qbase + M;
        float inv = 1.0f / lrow[j];
        size_t rowoff = ((size_t)(b * SEQ) + qrow) * HIDDEN + (size_t)h * HDIM;
        #pragma unroll
        for (int dt = 0; dt < 8; dt++)
            Oout[rowoff + dt * 16 + lo] = (__bf16)(O[dt][j] * inv);
    }
}

// ---------------------------------------------------------------- launch
extern "C" void kernel_launch(void* const* d_in, const int* in_sizes, int n_in,
                              void* d_out, int out_size, void* d_ws, size_t ws_size,
                              hipStream_t stream) {
    const int*   positions = (const int*)d_in[0];
    const float* X  = (const float*)d_in[1];
    const float* Wq = (const float*)d_in[2];
    const float* Wk = (const float*)d_in[3];
    const float* Wv = (const float*)d_in[4];
    const float* Wo = (const float*)d_in[5];
    float* out = (float*)d_out;

    const size_t R = (size_t)4096 * 4096;       // 16M elems per region
    __bf16* ws   = (__bf16*)d_ws;
    __bf16* Xb   = ws;                          // r0 (later aliased by Qa)
    __bf16* Wqb  = ws + 1 * R;
    __bf16* Wkb  = ws + 2 * R;
    __bf16* Wvb  = ws + 3 * R;
    __bf16* Wob  = ws + 4 * R;
    __bf16* Qraw = ws + 5 * R;                  // later aliased by attn out
    __bf16* Kraw = ws + 6 * R;
    __bf16* Vraw = ws + 7 * R;
    __bf16* Ka   = ws + 8 * R;
    __bf16* Vt   = ws + 9 * R;
    __bf16* Qa   = Xb;                          // X dead after QKV GEMMs
    __bf16* AO   = Qraw;                        // Qraw dead after rope

    const int n4 = (int)(R / 4);
    dim3 cblk((n4 + 255) / 256);
    cvt_f32_bf16<<<cblk, 256, 0, stream>>>((const float4*)X,  Xb,  n4);
    cvt_f32_bf16<<<cblk, 256, 0, stream>>>((const float4*)Wq, Wqb, n4);
    cvt_f32_bf16<<<cblk, 256, 0, stream>>>((const float4*)Wk, Wkb, n4);
    cvt_f32_bf16<<<cblk, 256, 0, stream>>>((const float4*)Wv, Wvb, n4);
    cvt_f32_bf16<<<cblk, 256, 0, stream>>>((const float4*)Wo, Wob, n4);

    dim3 gg(32, 32);   // 4096/128 x 4096/128
    gemm_bf16_kernel<false><<<gg, 256, 0, stream>>>(Xb, Wqb, Qraw, 4096, 4096, 4096);
    gemm_bf16_kernel<false><<<gg, 256, 0, stream>>>(Xb, Wkb, Kraw, 4096, 4096, 4096);
    gemm_bf16_kernel<false><<<gg, 256, 0, stream>>>(Xb, Wvb, Vraw, 4096, 4096, 4096);

    rope_kernel<<<32768, 256, 0, stream>>>(positions, Qraw, Kraw, Vraw, Qa, Ka, Vt);

    attn_kernel<<<1024, 256, 0, stream>>>(Qa, Ka, Vt, AO);

    gemm_bf16_kernel<true><<<gg, 256, 0, stream>>>(AO, Wob, out, 4096, 4096, 4096);
}
